// NoisyTopKRouter_6219112645447
// MI455X (gfx1250) — compile-verified, matrix operand order checked by reading, not run on hardware
//
#include <hip/hip_runtime.h>
#include <stdint.h>

// ---------------- problem constants (match reference) ----------------
#define H_DIM 4096
#define E_DIM 64

// ---------------- tiling ----------------
#define KC        256                       // K-chunk staged in LDS
#define KSTEPS    (KC / 32)                 // 8 wmma k-steps per chunk
#define NCHUNK    (H_DIM / KC)              // 16
#define ROW_BYTES (KC * 2)                  // 512 B of fp16 per weight row per chunk
#define ROW_STRIDE (ROW_BYTES + 16)         // 528 B: +16 pad => 4-bank advance, no conflicts
#define NROWS     (2 * E_DIM)               // gate rows then noise rows
#define BUF_BYTES (NROWS * ROW_STRIDE)      // 67584
#define SMEM_BYTES (2 * BUF_BYTES)          // 135168 (double buffer; epilogue reuses buf0)
#define EPI_RS    68                        // floats per epilogue row (64 + pad)
#define EPI_WAVE_BYTES (32 * EPI_RS * 4)    // 8704 B per wave (16 gate rows + 16 noise rows)

#define WAVES        4
#define THREADS      (WAVES * 32)
#define TOK_PER_WAVE 16
#define TOK_PER_BLOCK (WAVES * TOK_PER_WAVE)

typedef __attribute__((ext_vector_type(16))) _Float16 v16h;
typedef __attribute__((ext_vector_type(8)))  float    v8f;
typedef __attribute__((ext_vector_type(4)))  float    f32x4;
typedef __attribute__((ext_vector_type(4)))  unsigned u32x4;

union V16 {
    v16h     h;
    unsigned u32[8];
    u32x4    q[2];
};

// One-instruction packed fp32x2 -> fp16x2 (v_cvt_pk_rtz: universally available).
// fp16 (10-bit mantissa, RTZ) beats bf16 (7-bit) for this data; range fits:
// |hidden| ~ N(0,1), |w| <= 1/64, accumulation stays in fp32.
__device__ __forceinline__ unsigned pack_f16x2(float lo, float hi) {
    auto p = __builtin_amdgcn_cvt_pkrtz(lo, hi);  // __fp16 x2: D[15:0]=cvt(lo), D[31:16]=cvt(hi)
    return __builtin_bit_cast(unsigned, p);
}

// CDNA5 async global->LDS copy (ASYNCcnt tracked); inline asm is portable
// across both toolchains' differing builtin arities.
__device__ __forceinline__ void async_b128(unsigned lds_off, unsigned long long gaddr) {
    asm volatile("global_load_async_to_lds_b128 %0, %1, off"
                 :: "v"(lds_off), "v"(gaddr) : "memory");
}
__device__ __forceinline__ void wait_async0() {
    asm volatile("s_wait_asynccnt 0" ::: "memory");
}

// ------------- prep: fp32 weights -> fp16 in workspace, [2][E][H] -------------
__global__ void prep_weights(const float* __restrict__ wg,
                             const float* __restrict__ wn,
                             unsigned* __restrict__ ws32) {
    int i = blockIdx.x * blockDim.x + threadIdx.x;       // dword index (2 elems)
    if (i < (E_DIM * H_DIM) / 2) {
        ws32[i]                       = pack_f16x2(wg[2 * i], wg[2 * i + 1]);
        ws32[(E_DIM * H_DIM) / 2 + i] = pack_f16x2(wn[2 * i], wn[2 * i + 1]);
    }
}

// ------------- main: dual GEMM (fp16 WMMA, fp32 acc) + fused noisy-top2 epilogue -------------
__global__ void __launch_bounds__(THREADS)
router_main(const float* __restrict__ hidden,
            const float* __restrict__ noise,
            const unsigned short* __restrict__ wsb,   // fp16 [2][64][4096]
            float* __restrict__ out_gate,
            int*   __restrict__ out_idx,
            float* __restrict__ out_logits) {
    extern __shared__ unsigned char smem[];

    const int tid  = threadIdx.x;
    const int lane = tid & 31;
    const int wave = tid >> 5;
    const int mrow = lane & 15;   // token-row within wave tile / expert col within n-tile
    const int sel  = lane >> 4;   // K half-selector per 16-bit WMMA operand layout
    const int tb   = blockIdx.x * TOK_PER_BLOCK + wave * TOK_PER_WAVE;

    const float* hrow = hidden + (size_t)(tb + mrow) * H_DIM;
    const unsigned long long wsaddr = (unsigned long long)(uintptr_t)wsb;

    v8f accg[4], accn[4];
#pragma unroll
    for (int t = 0; t < 4; ++t) {
        accg[t] = (v8f){0.f,0.f,0.f,0.f,0.f,0.f,0.f,0.f};
        accn[t] = (v8f){0.f,0.f,0.f,0.f,0.f,0.f,0.f,0.f};
    }

    // stage one K-chunk of both weight matrices into an LDS buffer (async)
    auto stage = [&](int chunk, int buf) {
        const unsigned base = (unsigned)buf * BUF_BYTES;
        const unsigned long long kcb = (unsigned long long)chunk * (KC * 2);
#pragma unroll
        for (int i = 0; i < (NROWS * 32) / THREADS; ++i) {   // 32 x b128 per thread
            int idx = tid + i * THREADS;
            int row = idx >> 5;                              // 0..127 (gate rows, then noise)
            int seg = idx & 31;                              // 16-byte segment in row chunk
            unsigned lds_off = (unsigned)(uintptr_t)&smem[base + row * ROW_STRIDE + seg * 16];
            unsigned long long g = wsaddr + (unsigned long long)row * (H_DIM * 2) + kcb + seg * 16;
            async_b128(lds_off, g);
        }
    };

    auto compute = [&](int buf, int kc0) {
        const unsigned base = (unsigned)buf * BUF_BYTES;
#pragma unroll
        for (int ks = 0; ks < KSTEPS; ++ks) {
            // A: 16x32 fp16 tile, ISA layout: lanes 0-15 hold K {0..7,16..23},
            // lanes 16-31 hold K {8..15,24..31}  (sel picks the +8 shift)
            const float* ap = hrow + kc0 + ks * 32 + sel * 8;
            f32x4 a0 = __builtin_nontemporal_load((const f32x4*)(ap));       // single-use stream:
            f32x4 a1 = __builtin_nontemporal_load((const f32x4*)(ap + 4));   // keep L2 for weights
            f32x4 a2 = __builtin_nontemporal_load((const f32x4*)(ap + 16));
            f32x4 a3 = __builtin_nontemporal_load((const f32x4*)(ap + 20));
            V16 A;
            A.u32[0] = pack_f16x2(a0[0], a0[1]);
            A.u32[1] = pack_f16x2(a0[2], a0[3]);
            A.u32[2] = pack_f16x2(a1[0], a1[1]);
            A.u32[3] = pack_f16x2(a1[2], a1[3]);
            A.u32[4] = pack_f16x2(a2[0], a2[1]);
            A.u32[5] = pack_f16x2(a2[2], a2[3]);
            A.u32[6] = pack_f16x2(a3[0], a3[1]);
            A.u32[7] = pack_f16x2(a3[2], a3[3]);
#pragma unroll
            for (int t = 0; t < 4; ++t) {                    // 4 n-tiles cover E=64
                int e = t * 16 + mrow;                       // B col-major == w row e
                unsigned bg = base + e * ROW_STRIDE + ks * 64 + sel * 16;
                unsigned bn = base + (E_DIM + e) * ROW_STRIDE + ks * 64 + sel * 16;
                V16 Bg, Bn;
                Bg.q[0] = *(const u32x4*)&smem[bg];
                Bg.q[1] = *(const u32x4*)&smem[bg + 32];
                Bn.q[0] = *(const u32x4*)&smem[bn];
                Bn.q[1] = *(const u32x4*)&smem[bn + 32];
                accg[t] = __builtin_amdgcn_wmma_f32_16x16x32_f16(
                    false, A.h, false, Bg.h, (short)0, accg[t], false, false);
                accn[t] = __builtin_amdgcn_wmma_f32_16x16x32_f16(
                    false, A.h, false, Bn.h, (short)0, accn[t], false, false);
            }
        }
    };

    // double-buffered pipeline
    stage(0, 0);
#pragma unroll 1
    for (int c = 0; c < NCHUNK; ++c) {
        wait_async0();                // my async writes for chunk c are done
        __syncthreads();              // everyone's are; previous buffer fully consumed
        if (c + 1 < NCHUNK) stage(c + 1, (c + 1) & 1);
        compute(c & 1, c * KC);
    }
    __syncthreads();

    // ---- epilogue: park logits in (reused) LDS, one lane per token row ----
    float* epi = (float*)(smem + wave * EPI_WAVE_BYTES);
#pragma unroll
    for (int t = 0; t < 4; ++t) {
#pragma unroll
        for (int r = 0; r < 8; ++r) {                 // C layout: M = r + 8*sel, N = mrow
            int m = r + 8 * sel;
            int e = t * 16 + mrow;
            epi[m * EPI_RS + e]        = accg[t][r];
            epi[(16 + m) * EPI_RS + e] = accn[t][r];
        }
    }

    if (lane < 16) {
        const int token = tb + lane;
        const float* nrow = noise + (size_t)token * E_DIM;
        float*       gl   = epi + lane * EPI_RS;          // gate logits -> final logits
        const float* nl   = epi + (16 + lane) * EPI_RS;   // noise logits
        float* logit_out  = out_logits + (size_t)token * E_DIM;

        float vmax = -3.4e38f;
        float v1 = -3.4e38f, v2 = -3.4e38f;
        int   i1 = 0, i2 = 0;
        for (int e = 0; e < E_DIM; ++e) {
            float x   = nl[e];
            float sp  = (x > 20.f) ? x : log1pf(__expf(x));   // softplus
            float l   = gl[e] + nrow[e] * sp;                 // NOISE_STD = 1
            gl[e]     = l;
            logit_out[e] = l;
            vmax = fmaxf(vmax, l);
            if (l > v1)      { v2 = v1; i2 = i1; v1 = l; i1 = e; }
            else if (l > v2) { v2 = l;  i2 = e; }
        }
        float s = 0.f;
        for (int e = 0; e < E_DIM; ++e) s += __expf(gl[e] - vmax);
        float p1 = __expf(v1 - vmax) / s;
        float p2 = __expf(v2 - vmax) / s;
        float inv = 1.f / (p1 + p2 + 1e-9f);
        float w1 = p1 * inv, w2 = p2 * inv;

        float* gout = out_gate + (size_t)token * E_DIM;
        for (int e = 0; e < E_DIM; ++e)
            gout[e] = (e == i1) ? w1 : ((e == i2) ? w2 : 0.f);
        out_idx[2 * token]     = i1;
        out_idx[2 * token + 1] = i2;
    }
}

extern "C" void kernel_launch(void* const* d_in, const int* in_sizes, int n_in,
                              void* d_out, int out_size, void* d_ws, size_t ws_size,
                              hipStream_t stream) {
    (void)n_in; (void)out_size; (void)ws_size;
    const float* hidden = (const float*)d_in[0];   // [B,S,H] f32
    const float* noise  = (const float*)d_in[1];   // [B,S,E] f32
    const float* wg     = (const float*)d_in[2];   // [E,H]   f32
    const float* wn     = (const float*)d_in[3];   // [E,H]   f32
    unsigned* ws        = (unsigned*)d_ws;         // fp16 [2][E][H] -> 1 MB

    const int T = in_sizes[0] / H_DIM;             // B*S tokens (16384)

    float* out_gate = (float*)d_out;                                    // T*E f32
    int*   out_idx  = (int*)((float*)d_out + (size_t)T * E_DIM);        // T*2 i32
    float* out_log  = (float*)d_out + (size_t)T * E_DIM + (size_t)T * 2;// T*E f32

    prep_weights<<<((E_DIM * H_DIM) / 2 + 255) / 256, 256, 0, stream>>>(wg, wn, ws);
    router_main<<<T / TOK_PER_BLOCK, THREADS, SMEM_BYTES, stream>>>(
        hidden, noise, (const unsigned short*)ws, out_gate, out_idx, out_log);
}
